// NeuralTensorDiagLayer_12730283065646
// MI455X (gfx1250) — matrix operand order repaired and checked by reading, not compiled, not run on hardware
//
#include <hip/hip_runtime.h>
#include <math.h>

#define B_DIM 4096
#define D_DIM 2048
#define N_DIM 2048
#define K_TOT (2 * D_DIM)

#define BM 128
#define BN 128
#define BK 32
#define LDT 40  // padded LDS stride in bf16 elements (80 bytes = 64B row + 16B pad)

typedef __attribute__((ext_vector_type(16))) __bf16 v16bf;
typedef __attribute__((ext_vector_type(8)))  float   v8f;
typedef __attribute__((ext_vector_type(8)))  __bf16 bf16x8;
typedef __attribute__((ext_vector_type(4)))  __bf16 bf16x4;
typedef __attribute__((ext_vector_type(4)))  float   f32x4;
typedef unsigned int u32;
typedef __attribute__((ext_vector_type(4))) u32 u32x4;
typedef __attribute__((ext_vector_type(8))) int i32x8;
typedef __attribute__((ext_vector_type(4))) int i32x4;

union FragA { v16bf v; bf16x8 h[2]; };

#if defined(__has_builtin)
#if __has_builtin(__builtin_amdgcn_tensor_load_to_lds) && __has_builtin(__builtin_amdgcn_s_wait_tensorcnt)
#define HAVE_TDM 1
#endif
#endif
#ifndef HAVE_TDM
#define HAVE_TDM 0
#endif

static __device__ __forceinline__ __bf16 f2bf(float f) {
  // round-to-nearest-even f32 -> bf16
  unsigned u = __float_as_uint(f);
  u += 0x7fffu + ((u >> 16) & 1u);
  unsigned short s = (unsigned short)(u >> 16);
  __bf16 r;
  __builtin_memcpy(&r, &s, sizeof(r));
  return r;
}

// branch-free tanh: 1 - 2/(e^{2|x|}+1), sign restored
static __device__ __forceinline__ float fast_tanh(float x) {
  float ax = __builtin_fabsf(x);
  float e  = __expf(2.0f * ax);                  // v_exp_f32
  float r  = __builtin_amdgcn_rcpf(e + 1.0f);    // v_rcp_f32
  return __builtin_copysignf(1.0f - 2.0f * r, x);
}

// ---------------- kernel 0: zero colsum ----------------
__global__ void ntd_zero_kernel(float* __restrict__ colsum) {
  colsum[blockIdx.x * 256 + threadIdx.x] = 0.0f;
}

// ---------------- kernel 1: colsum[d] = sum_b e1[b,d]*e2[b,d] ----------------
__global__ void ntd_colsum_kernel(const float* __restrict__ e1,
                                  const float* __restrict__ e2,
                                  float* __restrict__ colsum) {
  const int d  = blockIdx.x * 256 + threadIdx.x;
  const int r0 = blockIdx.y * 256;
  float s = 0.0f;
  for (int r = 0; r < 256; ++r) {
    const size_t idx = (size_t)(r0 + r) * D_DIM + d;
    s += e1[idx] * e2[idx];
  }
  atomicAdd(&colsum[d], s);
}

// ---------------- kernel 2: diag[k] = dot(colsum, W[k,:])/(B*D) + b[k] ------
__global__ void ntd_diag_kernel(const float* __restrict__ colsum,
                                const float* __restrict__ W,
                                const float* __restrict__ bvec,
                                float* __restrict__ diagv) {
  const int wave = threadIdx.x >> 5;
  const int lane = threadIdx.x & 31;
  const int k = blockIdx.x * 8 + wave;
  float s = 0.0f;
  const float* wrow = W + (size_t)k * D_DIM;
  for (int d = lane; d < D_DIM; d += 32) s += colsum[d] * wrow[d];
  for (int off = 16; off > 0; off >>= 1) s += __shfl_down(s, off, 32);
  if (lane == 0)
    diagv[k] = s * (1.0f / ((float)B_DIM * (float)D_DIM)) + bvec[k];
}

// ---------------- prep A: Abf[m][0:2048]=bf16(e1), [2048:4096]=bf16(e2) -----
__global__ void ntd_convA_kernel(const float* __restrict__ e1,
                                 const float* __restrict__ e2,
                                 __bf16* __restrict__ Abf) {
  const int row = blockIdx.x;
  const int t = threadIdx.x;
#pragma unroll
  for (int i = 0; i < 2; ++i) {
    const int c = (t + 256 * i) * 4;
    f32x4 v1 = *(const f32x4*)(e1 + (size_t)row * D_DIM + c);
    f32x4 v2 = *(const f32x4*)(e2 + (size_t)row * D_DIM + c);
    bf16x4 h1, h2;
#pragma unroll
    for (int j = 0; j < 4; ++j) { h1[j] = f2bf(v1[j]); h2[j] = f2bf(v2[j]); }
    *(bf16x4*)(Abf + (size_t)row * K_TOT + c) = h1;
    *(bf16x4*)(Abf + (size_t)row * K_TOT + D_DIM + c) = h2;
  }
}

// ---------------- prep B: Vt[n][k] = bf16(V[k][n])  (2048 x 4096) ------------
__global__ void ntd_transV_kernel(const float* __restrict__ V,
                                  __bf16* __restrict__ Vt) {
  __shared__ float tile[32][33];
  const int tx = threadIdx.x & 31;
  const int ty = threadIdx.x >> 5;     // 32x8 threads
  const int n0 = blockIdx.x * 32;
  const int d0 = blockIdx.y * 32;
#pragma unroll
  for (int r = ty; r < 32; r += 8)
    tile[r][tx] = V[(size_t)(d0 + r) * N_DIM + n0 + tx];
  __syncthreads();
#pragma unroll
  for (int r = ty; r < 32; r += 8)
    Vt[(size_t)(n0 + r) * K_TOT + d0 + tx] = f2bf(tile[tx][r]);
}

// =====================  TDM double-buffered GEMM  ============================
#if HAVE_TDM
static __device__ __forceinline__ void tdm_issue(u32x4 g0, i32x8 g1) {
  i32x4 z4 = {0, 0, 0, 0};
#if __clang_major__ >= 23
  i32x8 z8 = {0, 0, 0, 0, 0, 0, 0, 0};
  __builtin_amdgcn_tensor_load_to_lds(g0, g1, z4, z4, z8, 0);
#else
  __builtin_amdgcn_tensor_load_to_lds(g0, g1, z4, z4, 0);
#endif
}

// D# group1: data_size=2B, pad every 16 DWORDs (64B row) by 4 DWORDs (16B),
// 2D tile tile_d0 x tile_d1, tensor dims/stride in elements.
static __device__ __forceinline__ i32x8 tdm_g1(u32 tdim0, u32 tdim1,
                                               u32 tile_d0, u32 tile_d1,
                                               u32 stride0) {
  i32x8 g;
  g[0] = (int)((1u << 16) |        // data_size = 1 -> 2 bytes
               (1u << 20) |        // pad_enable
               (3u << 22) |        // pad_interval: 16 DWORDs
               (3u << 25));        // pad_amount:   4 DWORDs
  g[1] = (int)((tdim0 & 0xffffu) << 16);
  g[2] = (int)(((tdim0 >> 16) & 0xffffu) | ((tdim1 & 0xffffu) << 16));
  g[3] = (int)(((tdim1 >> 16) & 0xffffu) | ((tile_d0 & 0xffffu) << 16));
  g[4] = (int)(tile_d1 & 0xffffu);  // tile_dim2 = 0 (2D)
  g[5] = (int)stride0;
  g[6] = 0;                         // stride0 hi, stride1 lo
  g[7] = 0;
  return g;
}

static __device__ __forceinline__ u32x4 tdm_g0(u32 lds_byte, unsigned long long ga) {
  u32x4 g;
  g[0] = 1u;                                  // count=1 (valid), user mode
  g[1] = lds_byte;                            // LDS byte address
  g[2] = (u32)ga;                             // global_addr[31:0]
  g[3] = (u32)((ga >> 32) & 0x01FFFFFFu) | (2u << 30);  // addr[56:32] | type=2
  return g;
}

__global__ __launch_bounds__(256) void ntd_gemm_tdm_kernel(
    const __bf16* __restrict__ Abf, const __bf16* __restrict__ Vt,
    const float* __restrict__ diagv, float* __restrict__ out) {
  // single LDS array so descriptor LDS offsets start at 0
  __shared__ __bf16 smem[2 * BM * LDT + 2 * BN * LDT];

  // The TDM engine writes smem through the integer address in the descriptor,
  // invisible to the compiler. Plant a never-taken store behind an opaque
  // scalar zero so LLVM cannot prove "smem is never stored" and fold the
  // fragment loads to undef. Accesses to smem stay plain C++ expressions so
  // address-space inference keeps them as ds_load_b128 (not flat/SYS).
  unsigned opaque_zero = 0;
  asm volatile("" : "+s"(opaque_zero));
  if (opaque_zero) smem[opaque_zero & 1023] = f2bf(0.0f);

  const int t    = threadIdx.x;
  const int lane = t & 31;
  const int wave = t >> 5;
  const int wm   = wave >> 1;
  const int wn   = wave & 1;
  const int m0   = blockIdx.y * BM;
  const int n0   = blockIdx.x * BN;
  const int hi   = lane >> 4;
  const int l15  = lane & 15;

  const bool leader = (__builtin_amdgcn_readfirstlane(wave) == 0);

  const i32x8 g1A = tdm_g1(K_TOT, B_DIM, BK, BM, K_TOT);
  const i32x8 g1B = tdm_g1(K_TOT, N_DIM, BK, BN, K_TOT);

  v8f acc[2][4];
#pragma unroll
  for (int im = 0; im < 2; ++im)
#pragma unroll
    for (int in = 0; in < 4; ++in)
#pragma unroll
      for (int j = 0; j < 8; ++j) acc[im][in][j] = 0.0f;

  const int steps = K_TOT / BK;

  if (leader) {
    tdm_issue(tdm_g0(0u,
                     (unsigned long long)(Abf + (size_t)m0 * K_TOT)),
              g1A);
    tdm_issue(tdm_g0((2u * BM * LDT) * 2u,
                     (unsigned long long)(Vt + (size_t)n0 * K_TOT)),
              g1B);
  }

  for (int step = 0; step < steps; ++step) {
    const int p = step & 1;
    if (leader) {
      if (step + 1 < steps) {
        const int q  = 1 - p;
        const int k1 = (step + 1) * BK;
        tdm_issue(tdm_g0((u32)(q * (BM * LDT)) * 2u,
                         (unsigned long long)(Abf + (size_t)m0 * K_TOT + k1)),
                  g1A);
        tdm_issue(tdm_g0((u32)(2 * BM * LDT + q * (BN * LDT)) * 2u,
                         (unsigned long long)(Vt + (size_t)n0 * K_TOT + k1)),
                  g1B);
        __builtin_amdgcn_s_wait_tensorcnt(2);  // current buffer's pair done
      } else {
        __builtin_amdgcn_s_wait_tensorcnt(0);
      }
    }
    __syncthreads();          // buffer p visible to all waves
    asm volatile("" ::: "memory");  // TDM wrote LDS: drop cached LDS values

    const __bf16* As = smem + p * (BM * LDT);
    const __bf16* Bs = smem + 2 * BM * LDT + p * (BN * LDT);

    FragA afr[2], bfr[4];
#pragma unroll
    for (int im = 0; im < 2; ++im) {
      const int r = wm * 32 + im * 16 + l15;
      afr[im].h[0] = *(const bf16x8*)(As + r * LDT + hi * 8);
      afr[im].h[1] = *(const bf16x8*)(As + r * LDT + 16 + hi * 8);
    }
#pragma unroll
    for (int in = 0; in < 4; ++in) {
      const int c = wn * 64 + in * 16 + l15;
      bfr[in].h[0] = *(const bf16x8*)(Bs + c * LDT + hi * 8);
      bfr[in].h[1] = *(const bf16x8*)(Bs + c * LDT + 16 + hi * 8);
    }
#pragma unroll
    for (int im = 0; im < 2; ++im)
#pragma unroll
      for (int in = 0; in < 4; ++in)
        acc[im][in] = __builtin_amdgcn_wmma_f32_16x16x32_bf16(
            false, afr[im].v, false, bfr[in].v,
            (short)0, acc[im][in], false, false);

    __syncthreads();  // done reading buffer p (next issue overwrites it)
  }

#pragma unroll
  for (int in = 0; in < 4; ++in) {
    const int gn = n0 + wn * 64 + in * 16 + l15;
    const float dg = diagv[gn];
#pragma unroll
    for (int im = 0; im < 2; ++im) {
      const int rbase = m0 + wm * 32 + im * 16 + hi * 8;
#pragma unroll
      for (int r = 0; r < 8; ++r)
        out[(size_t)(rbase + r) * N_DIM + gn] = fast_tanh(acc[im][in][r] + dg);
    }
  }
}
#endif  // HAVE_TDM

// =====================  fallback manual-copy GEMM  ===========================
__global__ __launch_bounds__(256) void ntd_gemm_kernel(
    const float* __restrict__ e1, const float* __restrict__ e2,
    const float* __restrict__ V,  const float* __restrict__ diagv,
    float* __restrict__ out) {
  __shared__ __bf16 Asm[BM * LDT];
  __shared__ __bf16 Bsm[BN * LDT];

  const int t    = threadIdx.x;
  const int lane = t & 31;
  const int wave = t >> 5;
  const int wm   = wave >> 1;
  const int wn   = wave & 1;
  const int m0   = blockIdx.y * BM;
  const int n0   = blockIdx.x * BN;
  const int hi   = lane >> 4;
  const int l15  = lane & 15;

  const int bn4 = (t & 31) * 4;
  const int bd0 = (t >> 5) * 4;

  v8f acc[2][4];
#pragma unroll
  for (int im = 0; im < 2; ++im)
#pragma unroll
    for (int in = 0; in < 4; ++in)
#pragma unroll
      for (int j = 0; j < 8; ++j) acc[im][in][j] = 0.0f;

  for (int k0 = 0; k0 < K_TOT; k0 += BK) {
    const float* asrc = (k0 < D_DIM) ? e1 : e2;
    const int kk = (k0 < D_DIM) ? k0 : (k0 - D_DIM);

    f32x4 areg[4];
    int arow[4], acol[4];
#pragma unroll
    for (int i = 0; i < 4; ++i) {
      const int cid = t + 256 * i;
      arow[i] = cid >> 3;
      acol[i] = (cid & 7) * 4;
      areg[i] = *(const f32x4*)(asrc + (size_t)(m0 + arow[i]) * D_DIM + kk + acol[i]);
    }
    f32x4 breg[4];
#pragma unroll
    for (int i = 0; i < 4; ++i)
      breg[i] = *(const f32x4*)(V + (size_t)(k0 + bd0 + i) * N_DIM + n0 + bn4);

    __syncthreads();

#pragma unroll
    for (int i = 0; i < 4; ++i) {
      bf16x4 h;
#pragma unroll
      for (int j = 0; j < 4; ++j) h[j] = f2bf(areg[i][j]);
      *(bf16x4*)(&Asm[arow[i] * LDT + acol[i]]) = h;
    }
#pragma unroll
    for (int j = 0; j < 4; ++j) {
      bf16x4 h;
#pragma unroll
      for (int i = 0; i < 4; ++i) h[i] = f2bf(breg[i][j]);
      *(bf16x4*)(&Bsm[(bn4 + j) * LDT + bd0]) = h;
    }

    __syncthreads();

    FragA afr[2], bfr[4];
#pragma unroll
    for (int im = 0; im < 2; ++im) {
      const int r = wm * 32 + im * 16 + l15;
      afr[im].h[0] = *(const bf16x8*)(&Asm[r * LDT + hi * 8]);
      afr[im].h[1] = *(const bf16x8*)(&Asm[r * LDT + 16 + hi * 8]);
    }
#pragma unroll
    for (int in = 0; in < 4; ++in) {
      const int c = wn * 64 + in * 16 + l15;
      bfr[in].h[0] = *(const bf16x8*)(&Bsm[c * LDT + hi * 8]);
      bfr[in].h[1] = *(const bf16x8*)(&Bsm[c * LDT + 16 + hi * 8]);
    }

#pragma unroll
    for (int im = 0; im < 2; ++im)
#pragma unroll
      for (int in = 0; in < 4; ++in)
        acc[im][in] = __builtin_amdgcn_wmma_f32_16x16x32_bf16(
            false, afr[im].v, false, bfr[in].v,
            (short)0, acc[im][in], false, false);
  }

#pragma unroll
  for (int in = 0; in < 4; ++in) {
    const int gn = n0 + wn * 64 + in * 16 + l15;
    const float dg = diagv[gn];
#pragma unroll
    for (int im = 0; im < 2; ++im) {
      const int rbase = m0 + wm * 32 + im * 16 + hi * 8;
#pragma unroll
      for (int r = 0; r < 8; ++r)
        out[(size_t)(rbase + r) * N_DIM + gn] = fast_tanh(acc[im][in][r] + dg);
    }
  }
}

extern "C" void kernel_launch(void* const* d_in, const int* in_sizes, int n_in,
                              void* d_out, int out_size, void* d_ws, size_t ws_size,
                              hipStream_t stream) {
  const float* e1 = (const float*)d_in[0];
  const float* e2 = (const float*)d_in[1];
  const float* W  = (const float*)d_in[2];
  const float* V  = (const float*)d_in[3];
  const float* bv = (const float*)d_in[4];
  float* out = (float*)d_out;

  float* colsum = (float*)d_ws;            // 2048 f32
  float* diagv  = colsum + D_DIM;          // 2048 f32

  ntd_zero_kernel<<<D_DIM / 256, 256, 0, stream>>>(colsum);
  ntd_colsum_kernel<<<dim3(D_DIM / 256, B_DIM / 256), 256, 0, stream>>>(e1, e2, colsum);
  ntd_diag_kernel<<<N_DIM / 8, 256, 0, stream>>>(colsum, W, bv, diagv);

  const size_t bf_bytes = sizeof(__bf16) *
      ((size_t)B_DIM * K_TOT + (size_t)N_DIM * K_TOT);
  const size_t need = 16384 + bf_bytes;

#if HAVE_TDM
  if (ws_size >= need) {
    __bf16* Abf = (__bf16*)((char*)d_ws + 16384);
    __bf16* Vt  = Abf + (size_t)B_DIM * K_TOT;
    ntd_convA_kernel<<<B_DIM, 256, 0, stream>>>(e1, e2, Abf);
    ntd_transV_kernel<<<dim3(N_DIM / 32, K_TOT / 32), 256, 0, stream>>>(V, Vt);
    ntd_gemm_tdm_kernel<<<dim3(N_DIM / BN, B_DIM / BM), 256, 0, stream>>>(
        Abf, Vt, diagv, out);
    return;
  }
#endif
  (void)need;
  ntd_gemm_kernel<<<dim3(N_DIM / BN, B_DIM / BM), 256, 0, stream>>>(
      e1, e2, V, diagv, out);
}
